// RNN_26542897889418
// MI455X (gfx1250) — matrix operand order repaired and checked by reading, not compiled, not run on hardware
//
#include <hip/hip_runtime.h>

#define B_ 16
#define T_ 4096
#define D_ 128
#define H_ 256

typedef __attribute__((ext_vector_type(16))) __bf16 v16bf;
typedef __attribute__((ext_vector_type(8)))  float  v8f;

union ABfrag { v16bf bf; uint4 q[2]; };
union CDfrag { v8f  v;  float f[8]; };

__device__ __forceinline__ unsigned short f2bf(float x) {
  // round-to-nearest-even f32 -> bf16
  unsigned u = __float_as_uint(x);
  u += 0x7FFFu + ((u >> 16) & 1u);
  return (unsigned short)(u >> 16);
}
__device__ __forceinline__ unsigned pack2(float a, float b) {
  return (unsigned)f2bf(a) | ((unsigned)f2bf(b) << 16);
}
__device__ __forceinline__ float fast_tanh(float x) {
  // tanh(x) = 1 - 2/(e^{2x}+1); exp2/rcp are TRANS ops, co-execute with WMMA
  float e = __builtin_amdgcn_exp2f(x * 2.8853900817779268f); // 2*log2(e)
  return 1.0f - 2.0f * __builtin_amdgcn_rcpf(e + 1.0f);
}

// ---------------------------------------------------------------------------
// Kernel 0a: convert W_x [H,D] and W_h [H,H] f32 -> bf16 in workspace
// ---------------------------------------------------------------------------
__global__ __launch_bounds__(256) void wconvert(const float* __restrict__ Wx,
                                                const float* __restrict__ Wh,
                                                unsigned short* __restrict__ wsx,
                                                unsigned short* __restrict__ wsh) {
  int i = blockIdx.x * 256 + threadIdx.x;
  if (i < H_ * D_) wsx[i] = f2bf(Wx[i]);
  if (i < H_ * H_) wsh[i] = f2bf(Wh[i]);
}

// ---------------------------------------------------------------------------
// Kernel 0b: convert X [B,T,D] f32 -> bf16 in workspace (one-time, BW-bound).
// Keeps all conversion VALU out of the GEMM's WMMA loop.
// ---------------------------------------------------------------------------
__global__ __launch_bounds__(256) void xconvert(const float* __restrict__ X,
                                                unsigned short* __restrict__ Xb) {
  int i = blockIdx.x * 256 + threadIdx.x;          // one float4 per thread
  const float4* src = (const float4*)X;
  uint2*        dst = (uint2*)Xb;
  float4 f = src[i];
  dst[i] = make_uint2(pack2(f.x, f.y), pack2(f.z, f.w));
}

// ---------------------------------------------------------------------------
// Kernel 1: xh = X @ W_x^T + b_x, written into d_out (staging; [B*T, H] f32)
// One 16x16 output tile per wave, K=128 as 4x wmma_f32_16x16x32_bf16.
// A and B fragments are straight global_load_b128 of pre-converted bf16.
// ---------------------------------------------------------------------------
__global__ __launch_bounds__(256) void x2h_gemm(const unsigned short* __restrict__ Xb,
                                                const unsigned short* __restrict__ Wx,
                                                const float* __restrict__ bx,
                                                float* __restrict__ out) {
  const int lane = threadIdx.x & 31;
  const int wave = threadIdx.x >> 5;
  const int half = lane >> 4;   // 0: lanes 0-15, 1: lanes 16-31
  const int ln   = lane & 15;
  const int g    = blockIdx.x * 8 + wave;   // global wave id
  const int mt   = g >> 4;                  // 0..4095  M-tile (rows of B*T)
  const int nt   = g & 15;                  // 0..15    N-tile (cols of H)
  const int col  = nt * 16 + ln;

  CDfrag c;
  float bias = bx[col];
#pragma unroll
  for (int r = 0; r < 8; ++r) c.f[r] = bias;

  const unsigned short* xrow = Xb + (size_t)(mt * 16 + ln) * D_; // A row = lane%16
  const unsigned short* wrow = Wx + (size_t)col * D_;            // B col n = W_x row n

#pragma unroll
  for (int k0 = 0; k0 < D_; k0 += 32) {
    ABfrag a, b;
    // A (16-bit, 16x32): lane<16 holds K k0+0..15 contiguous; lane>=16 K k0+16..31
    int ka = k0 + (half ? 16 : 0);
    a.q[0] = *(const uint4*)(xrow + ka);
    a.q[1] = *(const uint4*)(xrow + ka + 8);
    // B (16-bit, 32x16): lane<16 holds K k0+0..15 contiguous; lane>=16 K k0+16..31
    const uint4* wp = (const uint4*)(wrow + k0 + (half ? 16 : 0));
    b.q[0] = wp[0];
    b.q[1] = wp[1];
    c.v = __builtin_amdgcn_wmma_f32_16x16x32_bf16(false, a.bf, false, b.bf,
                                                  (short)0, c.v, false, false);
  }
  // D: VGPR r -> row m = r + 8*half, col = lane%16
#pragma unroll
  for (int r = 0; r < 8; ++r) {
    int m = mt * 16 + r + half * 8;
    out[(size_t)m * H_ + col] = c.f[r];
  }
}

// ---------------------------------------------------------------------------
// Kernel 2: sequential scan. One workgroup (8 waves). Wave w owns N-tiles
// {2w, 2w+1}. W_h bf16 fragments pinned in VGPRs (128 VGPRs/wave), h double-
// buffered in LDS as bf16 with +16B row padding (conflict-free ds_load_b128).
// ---------------------------------------------------------------------------
__global__ __launch_bounds__(256) void rnn_scan(float* __restrict__ out,
                                                const unsigned short* __restrict__ Wh,
                                                const float* __restrict__ bh) {
  __shared__ unsigned short hbuf[2][16][264];   // 16.5 KB, padded rows
  const int lane = threadIdx.x & 31;
  const int wave = threadIdx.x >> 5;
  const int half = lane >> 4;
  const int ln   = lane & 15;

  // Pin W_h B-fragments in registers: 2 tiles x 8 K-chunks x v16bf
  ABfrag Bf[2][8];
#pragma unroll
  for (int tile = 0; tile < 2; ++tile) {
    int col = wave * 32 + tile * 16 + ln;
    const unsigned short* wrow = Wh + (size_t)col * H_;
#pragma unroll
    for (int k = 0; k < 8; ++k) {
      const uint4* wp = (const uint4*)(wrow + k * 32 + (half ? 16 : 0));
      Bf[tile][k].q[0] = wp[0];
      Bf[tile][k].q[1] = wp[1];
    }
  }

  // h0 = 0
  for (int i = threadIdx.x; i < 16 * 264; i += 256)
    ((unsigned short*)hbuf)[i] = 0;
  __syncthreads();

  const float bias0 = bh[wave * 32 + ln];
  const float bias1 = bh[wave * 32 + 16 + ln];

  for (int t = 0; t < T_; ++t) {
    const int cur = t & 1, nxt = cur ^ 1;
#pragma unroll
    for (int tile = 0; tile < 2; ++tile) {
      const int col  = wave * 32 + tile * 16 + ln;
      const float bias = tile ? bias1 : bias0;
      CDfrag c;
      // C init = xh(b, t, col) + b_h  (xh sits in L2: 64MB < 192MB)
#pragma unroll
      for (int r = 0; r < 8; ++r) {
        int m = r + half * 8;                        // batch index
        c.f[r] = out[((size_t)m * T_ + t) * H_ + col] + bias;
      }
      // h_{t-1} @ W_h^T : K = 256 as 8 bf16 WMMAs, A from LDS
#pragma unroll
      for (int k = 0; k < 8; ++k) {
        ABfrag a;
        a.q[0] = *(const uint4*)&hbuf[cur][ln][k * 32 + half * 16];
        a.q[1] = *(const uint4*)&hbuf[cur][ln][k * 32 + half * 16 + 8];
        c.v = __builtin_amdgcn_wmma_f32_16x16x32_bf16(false, a.bf, false,
                                                      Bf[tile][k].bf,
                                                      (short)0, c.v, false, false);
      }
      // tanh, write h_t to d_out (f32) and next LDS buffer (bf16)
#pragma unroll
      for (int r = 0; r < 8; ++r) {
        int m = r + half * 8;
        float h = fast_tanh(c.f[r]);
        out[((size_t)m * T_ + t) * H_ + col] = h;
        hbuf[nxt][m][col] = f2bf(h);
      }
      // warm L2->L0 path for next step's xh
      if (t + 1 < T_)
        __builtin_prefetch(&out[((size_t)(half * 8) * T_ + (t + 1)) * H_ + col], 0, 1);
    }
    __syncthreads();
  }
}

// ---------------------------------------------------------------------------
extern "C" void kernel_launch(void* const* d_in, const int* in_sizes, int n_in,
                              void* d_out, int out_size, void* d_ws, size_t ws_size,
                              hipStream_t stream) {
  const float* X  = (const float*)d_in[0];  // [B,T,D]
  const float* Wx = (const float*)d_in[1];  // [H,D]
  const float* bx = (const float*)d_in[2];  // [H]
  const float* Wh = (const float*)d_in[3];  // [H,H]
  const float* bh = (const float*)d_in[4];  // [H]
  float* out = (float*)d_out;               // [B,T,H]

  // workspace layout: [X bf16: B*T*D][Wx bf16: H*D][Wh bf16: H*H]
  unsigned short* xb  = (unsigned short*)d_ws;
  unsigned short* wsx = xb + (size_t)B_ * T_ * D_;
  unsigned short* wsh = wsx + H_ * D_;

  wconvert<<<(H_ * H_ + 255) / 256, 256, 0, stream>>>(Wx, Wh, wsx, wsh);
  xconvert<<<(B_ * T_ * D_ / 4) / 256, 256, 0, stream>>>(X, xb);
  x2h_gemm<<<(B_ * T_ / 16) * 16 / 8, 256, 0, stream>>>(xb, wsx, bx, out);
  rnn_scan<<<1, 256, 0, stream>>>(out, wsh, bh);
}
